// STFC3Net_65627100283545
// MI455X (gfx1250) — matrix operand order repaired and checked by reading, not compiled, not run on hardware
//
#include <hip/hip_runtime.h>

// ---------------------------------------------------------------------------
// STFC pipeline for MI455X (gfx1250, wave32, WMMA).
// Dominant cost: 80x (1024^3) matmuls in the scan chain (~172 GFLOP) ->
// v_wmma_f32_16x16x32_f16 with f32 accumulation, double-buffered LDS tiles
// staged with GLOBAL_LOAD_ASYNC_TO_LDS_B128 (ASYNCcnt) when available.
// ---------------------------------------------------------------------------

typedef _Float16 v16h __attribute__((ext_vector_type(16)));
typedef _Float16 v8h  __attribute__((ext_vector_type(8)));
typedef float    v8f  __attribute__((ext_vector_type(8)));
typedef float    f4   __attribute__((ext_vector_type(4)));

// gcc-style vectors to match the async-LDS builtin prototype exactly
typedef int gv4i __attribute__((vector_size(16)));
typedef __attribute__((address_space(1))) gv4i as1_v4i;
typedef __attribute__((address_space(3))) gv4i as3_v4i;

#define INV_TEMP (1.0f / 0.07f)

#if defined(__HIP_DEVICE_COMPILE__) && \
    __has_builtin(__builtin_amdgcn_global_load_async_to_lds_b128) && \
    __has_builtin(__builtin_amdgcn_s_wait_asynccnt)
#define ASYNC_LDS 1
#else
#define ASYNC_LDS 0
#endif

// 16-byte global -> LDS copy. Async path goes straight to LDS (no VGPR
// round-trip), tracked by ASYNCcnt; fallback is load+ds_store.
// AS3 pointer: low 32 bits of a flat LDS address are the LDS offset
static __device__ __forceinline__ void copy16_g2l(_Float16* l, const _Float16* g) {
#if ASYNC_LDS
    __builtin_amdgcn_global_load_async_to_lds_b128(
        (as1_v4i*)(unsigned long long)(g),
        (as3_v4i*)(unsigned int)(unsigned long long)(l),
        0, 0);
#else
    *(f4*)l = *(const f4*)g;
#endif
}

static __device__ __forceinline__ void async_fence() {
#if ASYNC_LDS
    __builtin_amdgcn_s_wait_asynccnt(0);
#endif
}

// A-operand fragment (16x32, M x K) from an LDS tile with 32-half rows.
// ISA 16-bit A layout: lanes 0-15 hold K 0-7 & 16-23, lanes 16-31 hold
// K 8-15 & 24-31; row = row0 + (lane & 15).
static __device__ __forceinline__ v16h frag_a_from_lds(const _Float16* s, int row0, int lane) {
    int r  = row0 + (lane & 15);
    int kh = (lane & 16) ? 8 : 0;
    const _Float16* p = s + r * 32 + kh;
    v8h lo = *(const v8h*)(p);
    v8h hi = *(const v8h*)(p + 16);
    v16h o;
#pragma unroll
    for (int i = 0; i < 8; ++i) { o[i] = lo[i]; o[8 + i] = hi[i]; }
    return o;
}

// B-operand fragment (32x16, K x N) from an LDS tile holding B^T rows
// (32 halves of K per column). ISA 16-bit B layout: per lane one column
// N = col0 + (lane & 15); lanes 0-15 hold K 0-7 & 16-23, lanes 16-31 hold
// K 8-15 & 24-31 (same K order as the A fragment).
static __device__ __forceinline__ v16h frag_b_from_lds(const _Float16* s, int col0, int lane) {
    int c  = col0 + (lane & 15);
    int kb = (lane & 16) ? 8 : 0;
    const _Float16* p = s + c * 32 + kb;
    v8h lo = *(const v8h*)(p);
    v8h hi = *(const v8h*)(p + 16);
    v16h o;
#pragma unroll
    for (int i = 0; i < 8; ++i) { o[i] = lo[i]; o[8 + i] = hi[i]; }
    return o;
}

// C[m,n] = sum_k A[m,k] * Bt[n,k]   (both operands row-major over k)
// Block computes a 128x128 C tile; 8 waves, each wave 32(M) x 64(N).
// flags: bit0 = store f16, bit1 = store transposed (C^T).
// Per-z operand offsets: off = (z / zdiv) * Out + (z % zdiv) * In  (elements).
__global__ __launch_bounds__(256) void gemm_rr_wmma(
    const _Float16* __restrict__ A, const _Float16* __restrict__ Bt,
    void* __restrict__ C,
    int K, int lda, int ldb, int ldc,
    int zdiv,
    long long aOut, long long aIn,
    long long bOut, long long bIn,
    long long cOut, long long cIn,
    int flags)
{
    __shared__ __align__(16) _Float16 As[2][128 * 32];
    __shared__ __align__(16) _Float16 Bs[2][128 * 32];

    int z  = blockIdx.z;
    int zo = z / zdiv, zi = z % zdiv;
    A  += zo * aOut + zi * aIn;
    Bt += zo * bOut + zi * bIn;
    long long coff = zo * cOut + zi * cIn;

    int tid   = threadIdx.x;
    int lane  = tid & 31;
    int wid   = tid >> 5;
    int waveM = wid & 3;   // 0..3 -> 32-row strips
    int waveN = wid >> 2;  // 0..1 -> 64-col strips
    int blockM = blockIdx.x * 128;
    int blockN = blockIdx.y * 128;

    // Global->LDS staging: each thread owns a 16-half (32B) chunk.
    int lr = tid >> 1;            // row 0..127 of the tile
    int lk = (tid & 1) * 16;      // half-offset 0 or 16
    const _Float16* gA = A  + (long long)(blockM + lr) * lda + lk;
    const _Float16* gB = Bt + (long long)(blockN + lr) * ldb + lk;
    int soff = lr * 32 + lk;

    auto stage_pair = [&](int buf, const _Float16* ga, const _Float16* gb) {
        copy16_g2l(&As[buf][soff],     ga);
        copy16_g2l(&As[buf][soff] + 8, ga + 8);
        copy16_g2l(&Bs[buf][soff],     gb);
        copy16_g2l(&Bs[buf][soff] + 8, gb + 8);
    };

    v8f acc[2][4] = {};

    stage_pair(0, gA, gB);   // preload tile 0
    int cur = 0;
    for (int kk = 0; kk < K; kk += 32) {
        async_fence();        // this wave's async LDS writes done
        __syncthreads();      // all waves' writes visible, readers drained

        if (kk + 32 < K) {    // overlap next tile copy with this tile's math
            gA += 32; gB += 32;
            stage_pair(cur ^ 1, gA, gB);
            __builtin_prefetch(gA + 32, 0, 0);   // global_prefetch_b8
            __builtin_prefetch(gB + 32, 0, 0);
        }

        const _Float16* as = As[cur];
        const _Float16* bs = Bs[cur];
        v16h af[2], bf[4];
#pragma unroll
        for (int i = 0; i < 2; ++i) af[i] = frag_a_from_lds(as, waveM * 32 + i * 16, lane);
#pragma unroll
        for (int j = 0; j < 4; ++j) bf[j] = frag_b_from_lds(bs, waveN * 64 + j * 16, lane);
#pragma unroll
        for (int i = 0; i < 2; ++i)
#pragma unroll
            for (int j = 0; j < 4; ++j)
                acc[i][j] = __builtin_amdgcn_wmma_f32_16x16x32_f16(
                    false, af[i], false, bf[j], (short)0, acc[i][j], false, false);
        cur ^= 1;
    }

    // Epilogue. C/D layout: VGPR r, lanes 0-15 -> M=r, N=lane; lanes 16-31 -> M=r+8.
    int hi8 = (lane >> 4) * 8;
    int nc  = lane & 15;
    bool outF16 = (flags & 1) != 0;
    bool transC = (flags & 2) != 0;
#pragma unroll
    for (int i = 0; i < 2; ++i) {
#pragma unroll
        for (int j = 0; j < 4; ++j) {
            int mBase = blockM + waveM * 32 + i * 16 + hi8;
            int n     = blockN + waveN * 64 + j * 16 + nc;
#pragma unroll
            for (int r = 0; r < 8; ++r) {
                int m = mBase + r;
                long long idx = transC ? ((long long)n * ldc + m)
                                       : ((long long)m * ldc + n);
                float v = acc[i][j][r];
                if (outF16) ((_Float16*)C)[coff + idx] = (_Float16)v;
                else        ((float*)C)[coff + idx]    = v;
            }
        }
    }
}

// conv_w (128,3,8,8) f32 -> f16 flat (k = c*64+ky*8+kx already matches layout)
__global__ void cast_f32_to_f16_kernel(const float* __restrict__ s,
                                       _Float16* __restrict__ d, int n) {
    int i = blockIdx.x * 256 + threadIdx.x;
    if (i < n) d[i] = (_Float16)s[i];
}

// x (64,3,256,256) f32 -> patches (64,1024,192) f16, k = c*64 + ky*8 + kx
__global__ void im2col_kernel(const float* __restrict__ x, _Float16* __restrict__ patches) {
    int idx = blockIdx.x * 256 + threadIdx.x;       // 0..65535 (frame*1024 + pixel)
    int g  = idx >> 10;
    int p  = idx & 1023;
    int oy = p >> 5, ox = p & 31;
    const float* xf = x + (long long)g * 3 * 65536;
    _Float16* out = patches + (long long)idx * 192;
#pragma unroll
    for (int c = 0; c < 3; ++c)
#pragma unroll
        for (int ky = 0; ky < 8; ++ky)
#pragma unroll
            for (int kx = 0; kx < 8; ++kx)
                out[c * 64 + ky * 8 + kx] =
                    (_Float16)xf[c * 65536 + (oy * 8 + ky) * 256 + ox * 8 + kx];
}

// convout (64,1024,128) f32 -> relu -> L2norm over 128ch
// writes featseq f16 (frame,pixel,ch) and featN f32 NCHW (frame,ch,pixel)
__global__ void relu_norm_kernel(const float* __restrict__ convout,
                                 _Float16* __restrict__ featseq,
                                 float* __restrict__ featN) {
    int lane = threadIdx.x & 31, wid = threadIdx.x >> 5;
    long long row = (long long)blockIdx.x * 8 + wid;     // 65536 rows
    int g = (int)(row >> 10), p = (int)(row & 1023);
    const float* src = convout + row * 128;
    float v[4]; float ss = 0.f;
#pragma unroll
    for (int i = 0; i < 4; ++i) {
        v[i] = fmaxf(src[lane + i * 32], 0.f);
        ss += v[i] * v[i];
    }
#pragma unroll
    for (int o = 16; o > 0; o >>= 1) ss += __shfl_xor(ss, o, 32);
    float inv = 1.f / fmaxf(sqrtf(ss), 1e-12f);
#pragma unroll
    for (int i = 0; i < 4; ++i) {
        int c = lane + i * 32;
        float y = v[i] * inv;
        featseq[row * 128 + c] = (_Float16)y;
        featN[((long long)g * 128 + c) * 1024 + p] = y;
    }
}

// Affine grid + bilinear sample of featN fwd frames + L2 renorm -> fa f16 seq.
// One wave per output pixel; lane handles channels lane, lane+32, lane+64, lane+96.
__global__ void grid_sample_norm_kernel(const float* __restrict__ featN,
                                        const float* __restrict__ theta,
                                        _Float16* __restrict__ fa) {
    int lane = threadIdx.x & 31, wid = threadIdx.x >> 5;
    long long row = (long long)blockIdx.x * 8 + wid;     // 32768 rows (frame*1024+pixel)
    int fr = (int)(row >> 10);     // 0..31 = b*8 + t
    int p  = (int)(row & 1023);
    int h = p >> 5, w = p & 31;
    int b = fr >> 3, t = fr & 7;

    const float* th = theta + b * 6;
    float gx = -1.f + (2.f / 31.f) * (float)w;
    float gy = -1.f + (2.f / 31.f) * (float)h;
    float X = th[0] * gx + th[1] * gy + th[2];
    float Y = th[3] * gx + th[4] * gy + th[5];
    float ix = (X + 1.f) * 0.5f * 31.f;
    float iy = (Y + 1.f) * 0.5f * 31.f;
    float x0f = floorf(ix), y0f = floorf(iy);
    int x0 = (int)x0f, y0 = (int)y0f;
    int x1 = x0 + 1,   y1 = y0 + 1;
    float wx1 = ix - x0f, wx0 = 1.f - wx1;
    float wy1 = iy - y0f, wy0 = 1.f - wy1;
    float mx0 = (x0 >= 0 && x0 <= 31) ? 1.f : 0.f;
    float mx1 = (x1 >= 0 && x1 <= 31) ? 1.f : 0.f;
    float my0 = (y0 >= 0 && y0 <= 31) ? 1.f : 0.f;
    float my1 = (y1 >= 0 && y1 <= 31) ? 1.f : 0.f;
    float w00 = wy0 * wx0 * my0 * mx0;
    float w01 = wy0 * wx1 * my0 * mx1;
    float w10 = wy1 * wx0 * my1 * mx0;
    float w11 = wy1 * wx1 * my1 * mx1;
    int x0c = min(max(x0, 0), 31), x1c = min(max(x1, 0), 31);
    int y0c = min(max(y0, 0), 31), y1c = min(max(y1, 0), 31);
    int i00 = y0c * 32 + x0c, i01 = y0c * 32 + x1c;
    int i10 = y1c * 32 + x0c, i11 = y1c * 32 + x1c;

    const float* f = featN + (long long)(b * 16 + t) * 128 * 1024;  // fwd frame
    float v[4]; float ss = 0.f;
#pragma unroll
    for (int i = 0; i < 4; ++i) {
        const float* fc = f + (long long)(lane + i * 32) * 1024;
        float s = w00 * fc[i00] + w01 * fc[i01] + w10 * fc[i10] + w11 * fc[i11];
        v[i] = s; ss += s * s;
    }
#pragma unroll
    for (int o = 16; o > 0; o >>= 1) ss += __shfl_xor(ss, o, 32);
    float inv = 1.f / fmaxf(sqrtf(ss), 1e-12f);
#pragma unroll
    for (int i = 0; i < 4; ++i)
        fa[row * 128 + lane + i * 32] = (_Float16)(v[i] * inv);
}

// In-place softmax over rows of 1024 f16 with temperature 0.07. Wave per row.
__global__ void softmax_rows_kernel(_Float16* __restrict__ data, long long nrows) {
    int lane = threadIdx.x & 31, wid = threadIdx.x >> 5;
    long long row = (long long)blockIdx.x * 8 + wid;
    if (row >= nrows) return;
    _Float16* p = data + row * 1024;
    float v[32];
    float mx = -3.4e38f;
#pragma unroll
    for (int i = 0; i < 32; ++i) {
        v[i] = (float)p[lane + i * 32];
        mx = fmaxf(mx, v[i]);
    }
#pragma unroll
    for (int o = 16; o > 0; o >>= 1) mx = fmaxf(mx, __shfl_xor(mx, o, 32));
    float s = 0.f;
#pragma unroll
    for (int i = 0; i < 32; ++i) {
        v[i] = __expf((v[i] - mx) * INV_TEMP);
        s += v[i];
    }
#pragma unroll
    for (int o = 16; o > 0; o >>= 1) s += __shfl_xor(s, o, 32);
    float r = 1.f / s;
#pragma unroll
    for (int i = 0; i < 32; ++i) p[lane + i * 32] = (_Float16)(v[i] * r);
}

// Batched 1024x1024 f16 transpose (64x64 LDS tiles, block 64x4).
__global__ void transpose_f16_kernel(const _Float16* __restrict__ src,
                                     _Float16* __restrict__ dst,
                                     int zdiv,
                                     long long sOut, long long sIn,
                                     long long dOut, long long dIn) {
    __shared__ _Float16 tile[64][65];
    int z = blockIdx.z, zo = z / zdiv, zi = z % zdiv;
    src += zo * sOut + zi * sIn;
    dst += zo * dOut + zi * dIn;
    int bx = blockIdx.x * 64, by = blockIdx.y * 64;
    int tx = threadIdx.x;
    for (int j = threadIdx.y; j < 64; j += 4)
        tile[j][tx] = src[(long long)(by + j) * 1024 + bx + tx];
    __syncthreads();
    for (int j = threadIdx.y; j < 64; j += 4)
        dst[(long long)(bx + j) * 1024 + by + tx] = tile[tx][j];
}

extern "C" void kernel_launch(void* const* d_in, const int* in_sizes, int n_in,
                              void* d_out, int out_size, void* d_ws, size_t ws_size,
                              hipStream_t stream) {
    (void)in_sizes; (void)n_in; (void)out_size; (void)ws_size;
    const float* x     = (const float*)d_in[0];   // (4,16,3,256,256)
    const float* theta = (const float*)d_in[1];   // (4,2,3)
    const float* convw = (const float*)d_in[2];   // (128,3,8,8)
    float* outF = (float*)d_out;                  // (6,4,1024,1024) f32

    const long long M1 = 1024LL * 1024LL;   // one attention matrix (elements)
    const long long FS = 1024LL * 128LL;    // one feature frame   (elements)

    char* ws = (char*)d_ws;
    size_t off = 0;
    auto alloc = [&](size_t bytes) -> char* {
        char* p = ws + off;
        off += (bytes + 255) & ~(size_t)255;
        return p;
    };
    _Float16* patches = (_Float16*)alloc(64LL * 1024 * 192 * 2);  // im2col
    _Float16* wh      = (_Float16*)alloc(128LL * 192 * 2);        // weights f16
    float*    convout = (float*)   alloc(64LL * 1024 * 128 * 4);  // conv logits
    _Float16* featseq = (_Float16*)alloc(64LL * FS * 2);          // (B,TT,1024,128)
    float*    featN   = (float*)   alloc(64LL * FS * 4);          // NCHW normalized
    _Float16* fa      = (_Float16*)alloc(32LL * FS * 2);          // sampled fwd feats
    _Float16* AA      = (_Float16*)alloc(60LL * M1 * 2);          // (b,t=0..14)
    _Float16* Aal     = (_Float16*)alloc(28LL * M1 * 2);          // (b,t=0..6)
    _Float16* AAt     = (_Float16*)alloc(28LL * M1 * 2);          // AA[:,8:]^T
    _Float16* tmpT    = (_Float16*)alloc(4LL * M1 * 2);           // tmp^T per batch
    _Float16* tmA     = (_Float16*)alloc(4LL * M1 * 2);
    _Float16* tmB     = (_Float16*)alloc(4LL * M1 * 2);

    const long long ZBIG = 1LL << 30;  // zdiv "disabled"

    // 1) weights -> f16 ; im2col
    cast_f32_to_f16_kernel<<<dim3(96), dim3(256), 0, stream>>>(convw, wh, 128 * 192);
    im2col_kernel<<<dim3(256), dim3(256), 0, stream>>>(x, patches);

    // 2) conv as implicit GEMM: per frame C(1024x128) = patches(1024x192) . wh^T
    gemm_rr_wmma<<<dim3(8, 1, 64), dim3(256), 0, stream>>>(
        patches, wh, (void*)convout, 192, 192, 192, 128,
        (int)ZBIG, 0LL, 192LL * 1024, 0LL, 0LL, 0LL, 128LL * 1024, 0);

    // 3) relu + L2norm -> featseq (f16) + featN (f32 NCHW)
    relu_norm_kernel<<<dim3(8192), dim3(256), 0, stream>>>(convout, featseq, featN);

    // 4) grid sample + renorm -> fa
    grid_sample_norm_kernel<<<dim3(4096), dim3(256), 0, stream>>>(featN, theta, fa);

    // 5) gram matmuls (K=128). AA: z=b*15+t -> feat[b,t] . feat[b,t+1]^T
    gemm_rr_wmma<<<dim3(8, 8, 60), dim3(256), 0, stream>>>(
        featseq, featseq + FS, (void*)AA, 128, 128, 128, 1024,
        15, 16 * FS, FS, 16 * FS, FS, 15 * M1, M1, 1);
    // Aal: z=b*7+t -> fa[b,t] . feat[b,t+1]^T
    gemm_rr_wmma<<<dim3(8, 8, 28), dim3(256), 0, stream>>>(
        fa, featseq + FS, (void*)Aal, 128, 128, 128, 1024,
        7, 8 * FS, FS, 16 * FS, FS, 7 * M1, M1, 1);

    // 6) softmax (fused 1/TEMP), in place
    softmax_rows_kernel<<<dim3(7680), dim3(256), 0, stream>>>(AA, 61440LL);
    softmax_rows_kernel<<<dim3(3584), dim3(256), 0, stream>>>(Aal, 28672LL);

    // 7) transpose aa_dst = AA[:,8:] for use as right operands
    transpose_f16_kernel<<<dim3(16, 16, 28), dim3(64, 4), 0, stream>>>(
        AA + 8 * M1, AAt, 7, 15 * M1, M1, 7 * M1, M1);

    // 8) scan: tm0 = AA[:,7]; step i: tmp = tm @ aa_dst[i];
    //    tm' = aa_src[i] @ tmp; At = al_src[i] @ tmp (i>=1 -> out slot 6-i)
    _Float16* tmCur = nullptr;
    for (int i = 0; i < 7; ++i) {
        const _Float16* Ap; long long aIn;
        if (i == 0) { Ap = AA + 7 * M1; aIn = 15 * M1; }
        else        { Ap = tmCur;       aIn = M1; }
        // tmp^T = (tm @ AA[b,8+i])^T  -- write transposed so it is a Bt operand
        gemm_rr_wmma<<<dim3(8, 8, 4), dim3(256), 0, stream>>>(
            Ap, AAt + (long long)i * M1, (void*)tmpT, 1024, 1024, 1024, 1024,
            (int)ZBIG, 0LL, aIn, 0LL, 7 * M1, 0LL, M1, 3);
        // tm' = AA[b,6-i] @ tmp
        _Float16* tmNext = (i & 1) ? tmA : tmB;
        gemm_rr_wmma<<<dim3(8, 8, 4), dim3(256), 0, stream>>>(
            AA + (long long)(6 - i) * M1, tmpT, (void*)tmNext, 1024, 1024, 1024, 1024,
            (int)ZBIG, 0LL, 15 * M1, 0LL, M1, 0LL, M1, 1);
        // At = Aal[b,6-i] @ tmp -> output slot (6-i) in (6,B,1024,1024)
        if (i >= 1) {
            gemm_rr_wmma<<<dim3(8, 8, 4), dim3(256), 0, stream>>>(
                Aal + (long long)(6 - i) * M1, tmpT,
                (void*)(outF + (long long)(6 - i) * 4 * M1), 1024, 1024, 1024, 1024,
                (int)ZBIG, 0LL, 7 * M1, 0LL, M1, 0LL, M1, 0);
        }
        tmCur = tmNext;
    }
}
